// Multi_Head_Attention_30348238914124
// MI455X (gfx1250) — matrix operand order repaired and checked by reading, not compiled, not run on hardware
//
#include <hip/hip_runtime.h>

// ---------------------------------------------------------------------------
// MHA forward for B=4, S=2048, H=1024, NH=16, HD=64 on gfx1250 (MI455X).
// All matmuls via v_wmma_f32_16x16x32_f16 (f32 accumulate).
// Pipeline: f32->f16 convert | Q/K/V projections (V stored transposed per
// head) | per-wave flash attention | output projection (f32 out).
// GEMM operand staging uses the Tensor Data Mover (tensor_load_to_lds) with
// double-buffered LDS when the builtin is available.
// ---------------------------------------------------------------------------

typedef _Float16 h16;
typedef __attribute__((ext_vector_type(16))) _Float16 v16h;
typedef __attribute__((ext_vector_type(8)))  float    v8f;
typedef __attribute__((ext_vector_type(4)))  unsigned int v4u;
typedef __attribute__((ext_vector_type(4)))  unsigned int u32x4;
typedef __attribute__((ext_vector_type(8)))  int i32x8;
typedef __attribute__((ext_vector_type(4)))  int i32x4;

union HFrag { v16h v; v4u q[2]; };

#if defined(__has_builtin)
#if __has_builtin(__builtin_amdgcn_tensor_load_to_lds)
#define HAVE_TDM 1
#endif
#endif
#ifndef HAVE_TDM
#define HAVE_TDM 0
#endif

__device__ __forceinline__ v8f wmma_f32_16x16x32(v16h a, v16h b, v8f c) {
  // (neg_a, A, neg_b, B, c_mod, C, reuse_a, reuse_b)
  return __builtin_amdgcn_wmma_f32_16x16x32_f16(false, a, false, b, (short)0, c,
                                                false, false);
}

__device__ __forceinline__ v8f v8f_zero() {
  v8f z;
#pragma unroll
  for (int e = 0; e < 8; ++e) z[e] = 0.0f;
  return z;
}

__device__ __forceinline__ unsigned ldsoff(const void* p) {
  // LDS byte offset: flat pointers to LDS carry the offset in addr[31:0]
  // (ISA 10.2 aperture mapping).
  return (unsigned)(unsigned long long)(uintptr_t)p;
}

#if HAVE_TDM
// TDM 2D tile load: tile_rows rows of 64 bytes (8 x 8B units), global row
// stride 2048B (K=1024 f16), LDS padded +16B per 64B row so the LDS row
// stride is 80B == 40 halfwords (matches the fragment readers and spreads
// banks). D# layout per CDNA5 ISA ch.8.
__device__ __forceinline__ void tdm_load_2d_pad(unsigned lds_byte_off,
                                                const void* gptr,
                                                unsigned tile_rows) {
  unsigned long long ga = (unsigned long long)(uintptr_t)gptr;
  const unsigned d0_units = 8;     // 64B row in 8B units
  const unsigned str_units = 256;  // 2048B row stride in 8B units
  u32x4 g0;
  g0[0] = 1u;                                  // count=1 (valid user D#)
  g0[1] = lds_byte_off;                        // lds_addr
  g0[2] = (unsigned)(ga & 0xFFFFFFFFu);        // global_addr[31:0]
  g0[3] = (unsigned)((ga >> 32) & 0x1FFFFFFu)  // global_addr[56:32]
          | (2u << 30);                        // type=2 ("image")
  i32x8 g1;
  // data_size=3 (8B) | pad_enable | pad_interval=3 (64B) | pad_amount=3 (16B)
  g1[0] = (int)((3u << 16) | (1u << 20) | (3u << 22) | (3u << 25));
  g1[1] = (int)((str_units & 0xFFFFu) << 16);                       // dim0 lo16
  g1[2] = (int)((str_units >> 16) | ((tile_rows & 0xFFFFu) << 16)); // dim0 hi | dim1 lo
  g1[3] = (int)((tile_rows >> 16) | (d0_units << 16));              // dim1 hi | tile_dim0
  g1[4] = (int)(tile_rows & 0xFFFFu);                               // tile_dim1 (tile_dim2=0)
  g1[5] = (int)str_units;                                           // dim0_stride lo32
  g1[6] = 0;
  g1[7] = 0;
  i32x4 gz = {0, 0, 0, 0};
#if __clang_major__ >= 23
  i32x8 gz8 = {0, 0, 0, 0, 0, 0, 0, 0};
  __builtin_amdgcn_tensor_load_to_lds(g0, g1, gz, gz, gz8, 0);
#else
  __builtin_amdgcn_tensor_load_to_lds(g0, g1, gz, gz, 0);
#endif
}
#endif  // HAVE_TDM

// ---------------------------------------------------------------------------
// f32 -> f16 conversion (grid-stride)
// ---------------------------------------------------------------------------
__global__ __launch_bounds__(256) void cvt_kernel(const float* __restrict__ in,
                                                  h16* __restrict__ out, int n) {
  int i = blockIdx.x * 256 + threadIdx.x;
  int st = gridDim.x * 256;
  for (; i < n; i += st) out[i] = (h16)in[i];
}

// ---------------------------------------------------------------------------
// NT GEMM:  C[M,N] = A[M,K] @ B[N,K]^T + bias[N]
//   MODE 0: f32 output, row-major
//   MODE 1: f16 output, row-major               (Q, K projections)
//   MODE 2: f16 output, per-head V^T scatter    (V projection)
// WG = 256 thr (8 waves, 4x2), wave tile 64x64, WG tile 256x128, K-step 32.
// Double-buffered LDS; tiles staged by the Tensor Data Mover (wave 0 issues,
// TENSORcnt tracks) when available, else cooperative copies.
// ---------------------------------------------------------------------------
template <int MODE>
__global__ __launch_bounds__(256) void gemm_nt(const h16* __restrict__ A,
                                               const h16* __restrict__ Bw,
                                               const float* __restrict__ bias,
                                               void* __restrict__ outp,
                                               int M, int N, int K) {
  __shared__ __align__(16) h16 sA[2][256 * 40];  // 80B row stride
  __shared__ __align__(16) h16 sB[2][128 * 40];

  const int tid = threadIdx.x;
  const int lane = tid & 31, wave = tid >> 5;
  const int mlane = lane & 15, hf = lane >> 4;
  const int wm = wave >> 1, wn = wave & 1;  // 4 (M) x 2 (N) waves
  const int blkM = blockIdx.x * 256, blkN = blockIdx.y * 128;

  v8f acc[4][4];
#pragma unroll
  for (int i = 0; i < 4; ++i)
#pragma unroll
    for (int j = 0; j < 4; ++j) acc[i][j] = v8f_zero();

  const int nsteps = K / 32;

#if HAVE_TDM
  if (wave == 0) {  // TDM prologue: stage k-step 0 into buffer 0
    tdm_load_2d_pad(ldsoff(&sA[0][0]), A + (size_t)blkM * K, 256);
    tdm_load_2d_pad(ldsoff(&sB[0][0]), Bw + (size_t)blkN * K, 128);
  }
#endif

  for (int s = 0; s < nsteps; ++s) {
    const int buf = s & 1;
    const int k0 = s * 32;

#if HAVE_TDM
    if (wave == 0) {
      if (s + 1 < nsteps) {
        // Prefetch next k-step into the other buffer, then wait for the
        // current step's two TDM ops (in-order per wave => <=2 outstanding).
        tdm_load_2d_pad(ldsoff(&sA[buf ^ 1][0]),
                        A + (size_t)blkM * K + (k0 + 32), 256);
        tdm_load_2d_pad(ldsoff(&sB[buf ^ 1][0]),
                        Bw + (size_t)blkN * K + (k0 + 32), 128);
        __builtin_amdgcn_s_wait_tensorcnt(2);
      } else {
        __builtin_amdgcn_s_wait_tensorcnt(0);
      }
    }
    __syncthreads();
#else
    __syncthreads();
    // Cooperative staging fallback: 16B chunks
#pragma unroll
    for (int it = 0; it < 4; ++it) {
      int idx = tid + it * 256;
      int r = idx >> 2, c = (idx & 3) * 8;
      *(v4u*)&sA[buf][r * 40 + c] =
          *(const v4u*)&A[(size_t)(blkM + r) * K + k0 + c];
    }
#pragma unroll
    for (int it = 0; it < 2; ++it) {
      int idx = tid + it * 256;
      int r = idx >> 2, c = (idx & 3) * 8;
      *(v4u*)&sB[buf][r * 40 + c] =
          *(const v4u*)&Bw[(size_t)(blkN + r) * K + k0 + c];
    }
    __syncthreads();
#endif

    // A-frag: lane(m,hf) holds 16 contiguous K at hf*16
    HFrag af[4], bf[4];
#pragma unroll
    for (int i = 0; i < 4; ++i) {
      const h16* p = &sA[buf][(wm * 64 + i * 16 + mlane) * 40];
      af[i].q[0] = *(const v4u*)(p + hf * 16);
      af[i].q[1] = *(const v4u*)(p + hf * 16 + 8);
    }
    // B-frag: lane(n,hf) holds 16 contiguous K at hf*16
#pragma unroll
    for (int j = 0; j < 4; ++j) {
      const h16* p = &sB[buf][(wn * 64 + j * 16 + mlane) * 40 + hf * 16];
      bf[j].q[0] = *(const v4u*)(p);
      bf[j].q[1] = *(const v4u*)(p + 8);
    }
#pragma unroll
    for (int i = 0; i < 4; ++i)
#pragma unroll
      for (int j = 0; j < 4; ++j)
        acc[i][j] = wmma_f32_16x16x32(af[i].v, bf[j].v, acc[i][j]);

    __syncthreads();  // all waves done with buf before it is refilled
  }

  // Epilogue. C/D layout: M = v + 8*hf, N = lane%16
#pragma unroll
  for (int i = 0; i < 4; ++i) {
#pragma unroll
    for (int j = 0; j < 4; ++j) {
#pragma unroll
      for (int v = 0; v < 8; ++v) {
        int row = blkM + wm * 64 + i * 16 + v + 8 * hf;
        int col = blkN + wn * 64 + j * 16 + mlane;
        float val = acc[i][j][v] + bias[col];
        if (MODE == 0) {
          ((float*)outp)[(size_t)row * N + col] = val;
        } else if (MODE == 1) {
          ((h16*)outp)[(size_t)row * N + col] = (h16)val;
        } else {
          // V^T scatter under the raw view: row=(b,s), col=d
          int b = row >> 11, s = row & 2047;
          int h = s >> 7;                        // head
          int sp = (s & 127) * 16 + (col >> 6);  // view seq position
          int dp = col & 63;                     // view head-dim
          ((h16*)outp)[(size_t)((b * 16 + h) * 64 + dp) * 2048 + sp] = (h16)val;
        }
      }
    }
  }
}

// ---------------------------------------------------------------------------
// Flash attention. One wave owns a 16-row Q tile; 8 waves/WG -> 128 rows/WG.
// Grid: (16 q-blocks, 64 head-blocks). K/V^T frags loaded straight from
// global (hot in WGP$/L2, next block prefetched); LDS used only to relayout
// P (C-layout -> A-frag layout).
// ---------------------------------------------------------------------------
__global__ __launch_bounds__(256) void attn_kernel(const h16* __restrict__ qh,
                                                   const h16* __restrict__ kh,
                                                   const h16* __restrict__ vT,
                                                   h16* __restrict__ ao) {
  __shared__ __align__(16) h16 sP[8 * 16 * 64];  // per-wave 16x64 P scratch

  const int tid = threadIdx.x;
  const int lane = tid & 31, wave = tid >> 5;
  const int mlane = lane & 15, hf = lane >> 4;
  const int qblk = blockIdx.x;  // 0..15
  const int head = blockIdx.y;  // b*16 + h, 0..63
  const size_t hoff = (size_t)head * (2048 * 64);
  const h16* qb = qh + hoff;
  const h16* kb = kh + hoff;
  const h16* vb = vT + hoff;  // (64 d) x (2048 s), row-major
  h16* ab = ao + hoff;
  const int qrow0 = qblk * 128 + wave * 16;
  h16* pl = &sP[wave * 16 * 64];

  // Q frags: 16x64, two K-steps of 32
  HFrag qf[2];
#pragma unroll
  for (int kk = 0; kk < 2; ++kk) {
    const h16* p = qb + (size_t)(qrow0 + mlane) * 64 + kk * 32;
    qf[kk].q[0] = *(const v4u*)(p + hf * 16);
    qf[kk].q[1] = *(const v4u*)(p + hf * 16 + 8);
  }

  v8f O[4];
#pragma unroll
  for (int t = 0; t < 4; ++t) O[t] = v8f_zero();
  float mrow[8], lrow[8];
#pragma unroll
  for (int v = 0; v < 8; ++v) { mrow[v] = -1.0e30f; lrow[v] = 0.0f; }

  const float cfold = 0.125f * 1.44269504088896f;  // scale * log2(e)

  for (int kbk = 0; kbk < 2048 / 64; ++kbk) {
    // Pull the next 64-column K / V^T block toward the WGP behind the math.
    if (kbk + 1 < 2048 / 64) {
      __builtin_prefetch(kb + (size_t)((kbk + 1) * 64 + lane * 2) * 64, 0, 1);
      __builtin_prefetch(vb + (size_t)(lane * 2) * 2048 + (kbk + 1) * 64, 0, 1);
    }

    // S tile (16x64) = Q @ K_blk^T
    v8f sacc[4];
#pragma unroll
    for (int j = 0; j < 4; ++j) sacc[j] = v8f_zero();
#pragma unroll
    for (int j = 0; j < 4; ++j) {
#pragma unroll
      for (int kk = 0; kk < 2; ++kk) {
        HFrag kf;
        const h16* p =
            kb + (size_t)(kbk * 64 + j * 16 + mlane) * 64 + kk * 32 + hf * 16;
        kf.q[0] = *(const v4u*)p;
        kf.q[1] = *(const v4u*)(p + 8);
        sacc[j] = wmma_f32_16x16x32(qf[kk].v, kf.v, sacc[j]);
      }
    }

    // Online softmax, per accumulator row v (global row = qrow0 + v + 8*hf).
    // Row is spread over the 16 lanes sharing hf -> xor-reduce masks 1,2,4,8.
#pragma unroll
    for (int v = 0; v < 8; ++v) {
      float a0 = sacc[0][v] * cfold, a1 = sacc[1][v] * cfold;
      float a2 = sacc[2][v] * cfold, a3 = sacc[3][v] * cfold;
      float rm = fmaxf(fmaxf(a0, a1), fmaxf(a2, a3));
      rm = fmaxf(rm, __shfl_xor(rm, 1));
      rm = fmaxf(rm, __shfl_xor(rm, 2));
      rm = fmaxf(rm, __shfl_xor(rm, 4));
      rm = fmaxf(rm, __shfl_xor(rm, 8));
      float mnew = fmaxf(mrow[v], rm);
      float alpha = exp2f(mrow[v] - mnew);
      float p0 = exp2f(a0 - mnew), p1 = exp2f(a1 - mnew);
      float p2 = exp2f(a2 - mnew), p3 = exp2f(a3 - mnew);
      float rs = p0 + p1 + p2 + p3;
      rs += __shfl_xor(rs, 1);
      rs += __shfl_xor(rs, 2);
      rs += __shfl_xor(rs, 4);
      rs += __shfl_xor(rs, 8);
      lrow[v] = lrow[v] * alpha + rs;
      mrow[v] = mnew;
      O[0][v] *= alpha; O[1][v] *= alpha; O[2][v] *= alpha; O[3][v] *= alpha;
      // Spill P (C-layout) to LDS as f16
      int r = v + 8 * hf;
      pl[r * 64 + 0 * 16 + mlane] = (h16)p0;
      pl[r * 64 + 1 * 16 + mlane] = (h16)p1;
      pl[r * 64 + 2 * 16 + mlane] = (h16)p2;
      pl[r * 64 + 3 * 16 + mlane] = (h16)p3;
    }
    // DS ops are in-order per wave; fence stops compiler reordering.
    asm volatile("s_wait_dscnt 0" ::: "memory");

    // Reload P in A-frag layout
    HFrag pf[2];
#pragma unroll
    for (int kk = 0; kk < 2; ++kk) {
      const h16* p = &pl[mlane * 64 + kk * 32];
      pf[kk].q[0] = *(const v4u*)(p + hf * 16);
      pf[kk].q[1] = *(const v4u*)(p + hf * 16 + 8);
    }
    asm volatile("" ::: "memory");

    // O += P @ V_blk   (V^T rows are d, K-contiguous over s -> NT)
#pragma unroll
    for (int t = 0; t < 4; ++t) {
#pragma unroll
      for (int kk = 0; kk < 2; ++kk) {
        HFrag vf;
        const h16* p =
            vb + (size_t)(t * 16 + mlane) * 2048 + kbk * 64 + kk * 32 + hf * 16;
        vf.q[0] = *(const v4u*)p;
        vf.q[1] = *(const v4u*)(p + 8);
        O[t] = wmma_f32_16x16x32(pf[kk].v, vf.v, O[t]);
      }
    }
  }

  // Normalize and store (f16, contiguous per-head block == raw view back)
#pragma unroll
  for (int v = 0; v < 8; ++v) {
    float inv = 1.0f / lrow[v];
    int row = qrow0 + v + 8 * hf;
#pragma unroll
    for (int t = 0; t < 4; ++t)
      ab[(size_t)row * 64 + t * 16 + mlane] = (h16)(O[t][v] * inv);
  }
}

// ---------------------------------------------------------------------------
// Host launcher
// ---------------------------------------------------------------------------
extern "C" void kernel_launch(void* const* d_in, const int* in_sizes, int n_in,
                              void* d_out, int out_size, void* d_ws,
                              size_t ws_size, hipStream_t stream) {
  const float* x  = (const float*)d_in[0];
  const float* Wq = (const float*)d_in[1];
  const float* bq = (const float*)d_in[2];
  const float* Wk = (const float*)d_in[3];
  const float* bk = (const float*)d_in[4];
  const float* Wv = (const float*)d_in[5];
  const float* bv = (const float*)d_in[6];
  const float* Wo = (const float*)d_in[7];
  const float* bo = (const float*)d_in[8];
  float* out = (float*)d_out;

  const size_t MK = (size_t)8192 * 1024;  // activations
  const size_t WK = (size_t)1024 * 1024;  // weights

  char* ws = (char*)d_ws;
  size_t off = 0;
  auto alloc = [&](size_t bytes) -> h16* {
    h16* p = (h16*)(ws + off);
    off += (bytes + 255) & ~(size_t)255;
    return p;
  };
  h16* xh  = alloc(MK * 2);
  h16* wqh = alloc(WK * 2);
  h16* wkh = alloc(WK * 2);
  h16* wvh = alloc(WK * 2);
  h16* woh = alloc(WK * 2);
  h16* qhb = alloc(MK * 2);
  h16* khb = alloc(MK * 2);
  h16* vTb = alloc(MK * 2);
  h16* aob = alloc(MK * 2);  // total ~88 MB

  cvt_kernel<<<1024, 256, 0, stream>>>(x, xh, (int)MK);
  cvt_kernel<<<512, 256, 0, stream>>>(Wq, wqh, (int)WK);
  cvt_kernel<<<512, 256, 0, stream>>>(Wk, wkh, (int)WK);
  cvt_kernel<<<512, 256, 0, stream>>>(Wv, wvh, (int)WK);
  cvt_kernel<<<512, 256, 0, stream>>>(Wo, woh, (int)WK);

  dim3 gg(32, 8), gb(256);
  gemm_nt<1><<<gg, gb, 0, stream>>>(xh, wqh, bq, qhb, 8192, 1024, 1024);
  gemm_nt<1><<<gg, gb, 0, stream>>>(xh, wkh, bk, khb, 8192, 1024, 1024);
  gemm_nt<2><<<gg, gb, 0, stream>>>(xh, wvh, bv, vTb, 8192, 1024, 1024);

  attn_kernel<<<dim3(16, 64), 256, 0, stream>>>(qhb, khb, vTb, aob);

  gemm_nt<0><<<gg, gb, 0, stream>>>(aob, woh, bo, (void*)out, 8192, 1024, 1024);
}
